// MoE_Disentangled_25503515804129
// MI455X (gfx1250) — matrix operand order repaired and checked by reading, not compiled, not run on hardware
//
#include <hip/hip_runtime.h>
#include <stdint.h>

typedef __attribute__((ext_vector_type(16))) _Float16 v16h;
typedef __attribute__((ext_vector_type(8)))  float    v8f;

#define B_      4
#define N_      1024
#define D_      1024
#define E_      16
#define NH_     8
#define HD_     128
#define H_      2048
#define S_      1040           // E_ + N_
#define ROWS    (B_ * S_)      // 4160
#define TOPROWS (B_ * E_)      // 64
#define SPAD    1056           // 33 * 32 (padded key length)

#if __has_builtin(__builtin_amdgcn_global_load_async_to_lds_b128) && \
    __has_builtin(__builtin_amdgcn_s_wait_asynccnt)
#define HAVE_ASYNC_LDS 1
typedef int v4i_ __attribute__((vector_size(16)));
typedef __attribute__((address_space(1))) v4i_* gv4p;   // global int4*
typedef __attribute__((address_space(3))) v4i_* lv4p;   // LDS int4* (32-bit)
#else
#define HAVE_ASYNC_LDS 0
#endif

union Frag16 { v16h v; _Float16 h[16]; uint32_t u[8]; };

enum { MODE_F16 = 0, MODE_GELU_F16 = 1, MODE_RES_F32 = 2, MODE_W_F32 = 3 };

__device__ inline v8f wmma32(v16h a, v16h b, v8f c) {
  return __builtin_amdgcn_wmma_f32_16x16x32_f16(
      /*neg_a=*/false, a, /*neg_b=*/false, b,
      /*c_mod=*/(short)0, c, /*reuse_a=*/false, /*reuse_b=*/false);
}

// A-matrix fragment (16x32 f16): lane L holds row M = L%16; halves map to
// K = (L>=16)*16 + v*2 + slot  (contiguous layout, matches B fragments).
__device__ inline v16h load_a_f16(const _Float16* A, int row, int lda, int k0, int lane) {
  Frag16 f;
  const _Float16* base = A + (size_t)row * lda + k0 + ((lane >> 4) << 4);
  #pragma unroll
  for (int v = 0; v < 8; ++v) {
    int koff = v << 1;
    f.u[v] = *reinterpret_cast<const uint32_t*>(base + koff);
  }
  return f.v;
}

// One wave (wave32) computes a 64x16 output tile: 4 M-tiles of 16 against one
// N-tile of 16, K-loop in steps of 32 (v_wmma_f32_16x16x32_f16).
// B tile (32x16 fp32 weights) is staged to LDS with async-DMA double buffering
// when the toolchain exposes the gfx1250 async-to-LDS builtins.
template<int K, int LDB, int MODE>
__global__ __launch_bounds__(32)
void gemm64_wmma(const _Float16* __restrict__ A, size_t a_estride,
                 const float* __restrict__ Bw, size_t b_estride,
                 const float* __restrict__ bias, int bias_estride,
                 void* __restrict__ Cout, size_t c_estride,
                 const float* __restrict__ extra)  // residual (mode 2) or w[m][e] (mode 3)
{
  constexpr int NSTEPS = K / 32;
  int lane = threadIdx.x;
  int e  = blockIdx.z;
  int n0 = blockIdx.x * 16;
  int m0 = blockIdx.y * 64;
  int nl = lane & 15, hi = lane >> 4;
  const _Float16* Ae = A  + (size_t)e * a_estride;
  const float*    Be = Bw + (size_t)e * b_estride;

  v8f acc[4] = {};

#if HAVE_ASYNC_LDS
  __shared__ float btile[2][512];                    // 2 x (32 rows x 16 cols)
  const float* gsrc0 = Be + (size_t)lane * LDB + n0; // lane L streams tile row L
  auto issue_tile = [&](int buf, int k0) {
    const float* g = gsrc0 + (size_t)k0 * LDB;
    float* l = &btile[buf][lane * 16];
    #pragma unroll
    for (int i = 0; i < 4; ++i)
      __builtin_amdgcn_global_load_async_to_lds_b128(
          (gv4p)(uintptr_t)(g + i * 4),
          (lv4p)(uint32_t)(uintptr_t)(l + i * 4),   // flat LDS addr low 32b == LDS offset
          0, 0);
  };
  issue_tile(0, 0);
  for (int s = 0; s < NSTEPS; ++s) {
    int k0 = s * 32;
    if (s + 1 < NSTEPS) {                            // prefetch next tile via async DMA
      asm volatile("" ::: "memory");
      issue_tile((s + 1) & 1, k0 + 32);
      __builtin_amdgcn_s_wait_asynccnt(4);           // drain previous tile (in-order)
    } else {
      __builtin_amdgcn_s_wait_asynccnt(0);
    }
    asm volatile("" ::: "memory");
    Frag16 bf;                                       // B frag: K = hi*16+i, N = nl
    const float* lb = &btile[s & 1][hi * 256];
    #pragma unroll
    for (int i = 0; i < 16; ++i) bf.h[i] = (_Float16)lb[i * 16 + nl];
    #pragma unroll
    for (int mt = 0; mt < 4; ++mt) {
      v16h af = load_a_f16(Ae, m0 + mt * 16 + nl, K, k0, lane);
      acc[mt] = wmma32(af, bf.v, acc[mt]);
    }
  }
#else
  for (int s = 0; s < NSTEPS; ++s) {
    int k0 = s * 32;
    Frag16 bf;                                       // immediate-offset weight loads
    const float* wb = Be + (size_t)(k0 + hi * 16) * LDB + n0 + nl;
    #pragma unroll
    for (int i = 0; i < 16; ++i) bf.h[i] = (_Float16)wb[(size_t)i * LDB];
    if (s + 1 < NSTEPS)
      __builtin_prefetch(wb + (size_t)32 * LDB, 0, 1);
    #pragma unroll
    for (int mt = 0; mt < 4; ++mt) {
      v16h af = load_a_f16(Ae, m0 + mt * 16 + nl, K, k0, lane);
      acc[mt] = wmma32(af, bf.v, acc[mt]);
    }
  }
#endif

  int n   = n0 + nl;
  int hi8 = hi << 3;
  float bn = 0.f;
  if constexpr (MODE != MODE_F16) bn = bias[(size_t)e * bias_estride + n];
  #pragma unroll
  for (int mt = 0; mt < 4; ++mt) {
    #pragma unroll
    for (int r = 0; r < 8; ++r) {
      int m = m0 + mt * 16 + r + hi8;   // C layout: lanes<16 -> M=r, lanes>=16 -> M=8+r
      float val = acc[mt][r];
      if constexpr (MODE == MODE_F16) {
        ((_Float16*)Cout)[(size_t)e * c_estride + (size_t)m * LDB + n] = (_Float16)val;
      } else if constexpr (MODE == MODE_GELU_F16) {
        float x = val + bn;
        float g = 0.5f * x * (1.f + erff(x * 0.70710678118654752f));   // exact GELU
        ((_Float16*)Cout)[(size_t)e * c_estride + (size_t)m * LDB + n] = (_Float16)g;
      } else if constexpr (MODE == MODE_RES_F32) {
        float res = extra[(size_t)(m & 15) * LDB + n];  // x rows<E = expert_tokens_top[s]
        ((float*)Cout)[(size_t)m * LDB + n] = val + bn + res;
      } else {  // MODE_W_F32: per-expert weighted output, deterministic (no atomics)
        float wv = extra[m * E_ + e];
        ((float*)Cout)[(size_t)e * c_estride + (size_t)m * LDB + n] = (val + bn) * wv;
      }
    }
  }
}

// LayerNorm over D=1024. mode 0: concat(expert_tokens_top, inputs) source,
// writes full xn + compacted first-E rows. mode 1: fp32 source rows (x_top).
__global__ __launch_bounds__(256)
void ln_kernel(const float* __restrict__ inputs, const float* __restrict__ ett,
               const float* __restrict__ gamma, const float* __restrict__ beta,
               const float* __restrict__ x_src,
               _Float16* __restrict__ xn_out, _Float16* __restrict__ xn_top_out,
               int mode)
{
  __shared__ float s1[256], s2[256];
  int tid = threadIdx.x, row = blockIdx.x;
  const float* src;
  int b = 0, sidx = 0;
  if (mode == 0) {
    b = row / S_; sidx = row - b * S_;
    src = (sidx < E_) ? (ett + (size_t)sidx * D_)
                      : (inputs + ((size_t)b * N_ + (sidx - E_)) * D_);
  } else {
    src = x_src + (size_t)row * D_;
  }
  float x[4], sum = 0.f, sq = 0.f;
  #pragma unroll
  for (int i = 0; i < 4; ++i) { x[i] = src[tid + i * 256]; sum += x[i]; sq += x[i] * x[i]; }
  s1[tid] = sum; s2[tid] = sq; __syncthreads();
  for (int off = 128; off > 0; off >>= 1) {
    if (tid < off) { s1[tid] += s1[tid + off]; s2[tid] += s2[tid + off]; }
    __syncthreads();
  }
  float mean = s1[0] * (1.f / D_);
  float var  = s2[0] * (1.f / D_) - mean * mean;
  float inv  = rsqrtf(var + 1e-5f);
  _Float16* dst  = xn_out + (size_t)row * D_;
  _Float16* dst2 = (mode == 0 && sidx < E_) ? (xn_top_out + ((size_t)b * E_ + sidx) * D_) : nullptr;
  #pragma unroll
  for (int i = 0; i < 4; ++i) {
    int d = tid + i * 256;
    float y = (x[i] - mean) * inv * gamma[d] + beta[d];
    dst[d] = (_Float16)y;
    if (dst2) dst2[d] = (_Float16)y;
  }
}

// One wave per (batch, head). Scores (16x1040) via WMMA into LDS, serial
// softmax, then P @ V via WMMA with zero-padded key tail.
__global__ __launch_bounds__(32)
void attn_kernel(const _Float16* __restrict__ q,   // 64 x 1024
                 const _Float16* __restrict__ kv,  // 4160 x 2048 (k | v)
                 _Float16* __restrict__ o)         // 64 x 1024
{
  extern __shared__ char smem[];
  float*    sc = (float*)smem;                       // 16 x SPAD fp32
  _Float16* pb = (_Float16*)(smem + 16 * SPAD * 4);  // 16 x SPAD f16
  int lane = threadIdx.x;
  int b = blockIdx.x >> 3, h = blockIdx.x & 7;
  int nl = lane & 15, hi = lane >> 4;
  const float scale = 0.08838834764831845f;          // HD^-0.5

  v16h aq[4];
  #pragma unroll
  for (int kk = 0; kk < 4; ++kk)
    aq[kk] = load_a_f16(q, b * E_ + nl, D_, h * HD_ + kk * 32, lane);

  for (int kt = 0; kt < 65; ++kt) {                  // 1040 = 65 key tiles of 16
    v8f c = {};
    #pragma unroll
    for (int kk = 0; kk < 4; ++kk) {                 // K = HD = 128
      Frag16 bf;
      const _Float16* base =
          kv + (size_t)(b * S_ + kt * 16 + nl) * (2 * D_) + h * HD_ + kk * 32 + hi * 16;
      #pragma unroll
      for (int u = 0; u < 8; ++u) bf.u[u] = *reinterpret_cast<const uint32_t*>(base + u * 2);
      c = wmma32(aq[kk], bf.v, c);
    }
    #pragma unroll
    for (int r = 0; r < 8; ++r) sc[(r + hi * 8) * SPAD + kt * 16 + nl] = c[r] * scale;
  }
  #pragma unroll
  for (int r = 0; r < 8; ++r) pb[(r + hi * 8) * SPAD + 1040 + nl] = (_Float16)0.f; // pad
  __syncthreads();

  if (lane < 16) {                                   // softmax, one row per lane
    int row = lane;
    float mx = -3.4e38f;
    for (int j = 0; j < S_; ++j) mx = fmaxf(mx, sc[row * SPAD + j]);
    float s = 0.f;
    for (int j = 0; j < S_; ++j) s += __expf(sc[row * SPAD + j] - mx);
    float rinv = 1.f / s;
    for (int j = 0; j < S_; ++j)
      pb[row * SPAD + j] = (_Float16)(__expf(sc[row * SPAD + j] - mx) * rinv);
  }
  __syncthreads();

  for (int nt = 0; nt < 8; ++nt) {                   // o: 16 x 128, K over keys
    v8f c = {};
    const _Float16* vbase = kv + (size_t)(b * S_ + hi * 16) * (2 * D_)
                            + D_ + h * HD_ + nt * 16 + nl;
    for (int ks = 0; ks < 32; ++ks) {                // full key steps: imm offsets
      v16h ap = load_a_f16(pb, nl, SPAD, ks * 32, lane);
      Frag16 bf;
      const _Float16* vb = vbase + (size_t)ks * 32 * (2 * D_);
      #pragma unroll
      for (int t = 0; t < 16; ++t) bf.h[t] = vb[(size_t)t * (2 * D_)];
      c = wmma32(ap, bf.v, c);
    }
    {                                                // tail (keys 1024..1055, clamp OOB)
      v16h ap = load_a_f16(pb, nl, SPAD, 32 * 32, lane);
      Frag16 bf;
      #pragma unroll
      for (int t = 0; t < 16; ++t) {
        int key = 32 * 32 + hi * 16 + t;
        key = key < S_ ? key : (S_ - 1);             // p==0 there, value irrelevant
        bf.h[t] = kv[(size_t)(b * S_ + key) * (2 * D_) + D_ + h * HD_ + nt * 16 + nl];
      }
      c = wmma32(ap, bf.v, c);
    }
    #pragma unroll
    for (int r = 0; r < 8; ++r)
      o[(size_t)(b * E_ + r + hi * 8) * D_ + h * HD_ + nt * 16 + nl] = (_Float16)c[r];
  }
}

// Router: scores[m][e] = xn2[m] . met[e], top-2 (lax.top_k tie-break: lower idx), w = 0.5.
__global__ __launch_bounds__(256)
void scores_topk_kernel(const _Float16* __restrict__ xn2, const float* __restrict__ met,
                        float* __restrict__ w)
{
  __shared__ float red[256];
  __shared__ float sc[E_];
  __shared__ int ii[2];
  int tid = threadIdx.x, m = blockIdx.x;
  for (int e = 0; e < E_; ++e) {
    float p = 0.f;
    for (int d = tid; d < D_; d += 256)
      p += (float)xn2[(size_t)m * D_ + d] * met[(size_t)e * D_ + d];
    red[tid] = p; __syncthreads();
    for (int off = 128; off > 0; off >>= 1) {
      if (tid < off) red[tid] += red[tid + off];
      __syncthreads();
    }
    if (tid == 0) sc[e] = red[0];
    __syncthreads();
  }
  if (tid == 0) {
    int i1 = 0;
    for (int e = 1; e < E_; ++e) if (sc[e] > sc[i1]) i1 = e;
    int i2 = -1;
    for (int e = 0; e < E_; ++e) { if (e == i1) continue; if (i2 < 0 || sc[e] > sc[i2]) i2 = e; }
    ii[0] = i1; ii[1] = i2;
  }
  __syncthreads();
  if (tid < E_) w[m * E_ + tid] = (tid == ii[0] || tid == ii[1]) ? 0.5f : 0.f;
}

// Final: x = x_top + sum_e moe[e]; emit expert_features + confidence.
__global__ __launch_bounds__(256)
void out_ef_kernel(const float* __restrict__ xtop, const float* __restrict__ moe,
                   const float* __restrict__ Wg, const float* __restrict__ bg,
                   float* __restrict__ out)
{
  __shared__ float red[256];
  int tid = threadIdx.x, m = blockIdx.x;
  float p = 0.f;
  for (int d = tid; d < D_; d += 256) {
    float v = xtop[(size_t)m * D_ + d];
    #pragma unroll
    for (int e = 0; e < E_; ++e)
      v += moe[(size_t)e * TOPROWS * D_ + (size_t)m * D_ + d];
    out[(size_t)m * D_ + d] = v;
    p += v * Wg[d];
  }
  red[tid] = p; __syncthreads();
  for (int off = 128; off > 0; off >>= 1) {
    if (tid < off) red[tid] += red[tid + off];
    __syncthreads();
  }
  if (tid == 0)
    out[(size_t)B_ * E_ * D_ + m] = 1.f / (1.f + expf(-(red[0] + bg[0])));
}

__global__ void fused_kernel(const float* __restrict__ inputs, float* __restrict__ out)
{
  int i = blockIdx.x * blockDim.x + threadIdx.x;
  if (i >= B_ * D_) return;
  int b = i / D_, d = i - b * D_;
  out[(size_t)B_ * E_ * D_ + B_ * E_ + i] =
      0.5f * inputs[(size_t)b * N_ * D_ + d] + 0.5f * inputs[((size_t)b * N_ + 1) * D_ + d];
}

extern "C" void kernel_launch(void* const* d_in, const int* in_sizes, int n_in,
                              void* d_out, int out_size, void* d_ws, size_t ws_size,
                              hipStream_t stream) {
  (void)in_sizes; (void)n_in; (void)out_size; (void)ws_size;
  const float* inputs = (const float*)d_in[0];
  const float* ett    = (const float*)d_in[1];
  const float* ln1_s  = (const float*)d_in[2];
  const float* ln1_b  = (const float*)d_in[3];
  const float* ln2_s  = (const float*)d_in[4];
  const float* ln2_b  = (const float*)d_in[5];
  const float* Wq     = (const float*)d_in[6];
  const float* Wkv    = (const float*)d_in[7];
  const float* Wproj  = (const float*)d_in[8];
  const float* bproj  = (const float*)d_in[9];
  const float* met    = (const float*)d_in[10];
  const float* W1     = (const float*)d_in[11];
  const float* b1     = (const float*)d_in[12];
  const float* W2     = (const float*)d_in[13];
  const float* b2     = (const float*)d_in[14];
  const float* W3     = (const float*)d_in[15];
  const float* b3     = (const float*)d_in[16];
  const float* Wg     = (const float*)d_in[17];
  const float* bg     = (const float*)d_in[18];
  float* out = (float*)d_out;

  char* ws = (char*)d_ws;
  size_t off = 0;
  auto alloc = [&](size_t bytes) { size_t o = off; off = (off + bytes + 255) & ~(size_t)255; return o; };
  _Float16* xn    = (_Float16*)(ws + alloc((size_t)ROWS * D_ * 2));       // 8.5 MB
  _Float16* kvb   = (_Float16*)(ws + alloc((size_t)ROWS * 2 * D_ * 2));   // 17 MB
  _Float16* xntop = (_Float16*)(ws + alloc((size_t)TOPROWS * D_ * 2));
  _Float16* qb    = (_Float16*)(ws + alloc((size_t)TOPROWS * D_ * 2));
  _Float16* ob    = (_Float16*)(ws + alloc((size_t)TOPROWS * D_ * 2));
  _Float16* xn2   = (_Float16*)(ws + alloc((size_t)TOPROWS * D_ * 2));
  float*    xtop  = (float*)   (ws + alloc((size_t)TOPROWS * D_ * 4));
  float*    wbuf  = (float*)   (ws + alloc((size_t)TOPROWS * E_ * 4));
  _Float16* h1    = (_Float16*)(ws + alloc((size_t)E_ * TOPROWS * H_ * 2)); // 4 MB
  _Float16* h2    = (_Float16*)(ws + alloc((size_t)E_ * TOPROWS * H_ * 2)); // 4 MB
  float*    moe   = (float*)   (ws + alloc((size_t)E_ * TOPROWS * D_ * 4)); // 4 MB

  // 1) LN1 over all 4160 rows (concat source) -> xn (f16) + compact top rows
  ln_kernel<<<ROWS, 256, 0, stream>>>(inputs, ett, ln1_s, ln1_b, nullptr, xn, xntop, 0);
  // 2) KV = xn @ Wkv  (M=4160, K=1024, N=2048) -> f16
  gemm64_wmma<1024, 2048, MODE_F16><<<dim3(2 * D_ / 16, ROWS / 64, 1), 32, 0, stream>>>(
      xn, 0, Wkv, 0, nullptr, 0, kvb, 0, nullptr);
  // 3) Q (only first-E rows) = xntop @ Wq
  gemm64_wmma<1024, 1024, MODE_F16><<<dim3(D_ / 16, 1, 1), 32, 0, stream>>>(
      xntop, 0, Wq, 0, nullptr, 0, qb, 0, nullptr);
  // 4) Attention: one wave per (b, h); 101 KB LDS (scores + probs)
  attn_kernel<<<B_ * NH_, 32, 16 * SPAD * 4 + 16 * SPAD * 2, stream>>>(qb, kvb, ob);
  // 5) x_top = expert_tokens_top + o @ Wproj + bproj   (fp32)
  gemm64_wmma<1024, 1024, MODE_RES_F32><<<dim3(D_ / 16, 1, 1), 32, 0, stream>>>(
      ob, 0, Wproj, 0, bproj, 0, xtop, 0, ett);
  // 6) LN2 on the 64 top rows
  ln_kernel<<<TOPROWS, 256, 0, stream>>>(nullptr, nullptr, ln2_s, ln2_b, xtop, xn2, nullptr, 1);
  // 7) Router scores + top-2 weights
  scores_topk_kernel<<<TOPROWS, 256, 0, stream>>>(xn2, met, wbuf);
  // 8) h1 = gelu(xn2 @ W1[e] + b1[e])   (16 experts x 128 N-tiles)
  gemm64_wmma<1024, 2048, MODE_GELU_F16><<<dim3(H_ / 16, 1, E_), 32, 0, stream>>>(
      xn2, 0, W1, (size_t)D_ * H_, b1, H_, h1, (size_t)TOPROWS * H_, nullptr);
  // 9) h2 = gelu(h1 @ W2[e] + b2[e])
  gemm64_wmma<2048, 2048, MODE_GELU_F16><<<dim3(H_ / 16, 1, E_), 32, 0, stream>>>(
      h1, (size_t)TOPROWS * H_, W2, (size_t)H_ * H_, b2, H_, h2, (size_t)TOPROWS * H_, nullptr);
  // 10) moe[e] = (h2 @ W3[e] + b3[e]) * w[:, e]   (deterministic per-expert store)
  gemm64_wmma<2048, 1024, MODE_W_F32><<<dim3(D_ / 16, 1, E_), 32, 0, stream>>>(
      h2, (size_t)TOPROWS * H_, W3, (size_t)H_ * D_, b3, D_, moe, (size_t)TOPROWS * D_, wbuf);
  // 11) expert_features + confidence, 12) fused
  out_ef_kernel<<<TOPROWS, 256, 0, stream>>>(xtop, moe, Wg, bg, out);
  fused_kernel<<<(B_ * D_ + 255) / 256, 256, 0, stream>>>(inputs, out);
}
